// SelfAttention_55937654063304
// MI455X (gfx1250) — matrix operand order repaired and checked by reading, not compiled, not run on hardware
//
#include <hip/hip_runtime.h>
#include <math.h>

// CDNA5 / gfx1250: wave32, WMMA 16x16x32 f16 -> f32 accumulate.
typedef __attribute__((ext_vector_type(16))) _Float16 v16h;
typedef __attribute__((ext_vector_type(8)))  float    v8f;
typedef __attribute__((ext_vector_type(4)))  int      v4i;
typedef __attribute__((ext_vector_type(8)))  int      v8i;
typedef __attribute__((ext_vector_type(4)))  unsigned v4u;

#define T_SEQ  4096
#define N_EMBD 768
#define HEAD   64
#define NBATCH 4
#define QT_PER_BLK 4   // query tiles (waves) per flash block

#ifndef __has_builtin
#define __has_builtin(x) 0
#endif
#if __has_builtin(__builtin_amdgcn_global_load_async_to_lds_b128)
#define HAVE_ASYNC_B128 1
#else
#define HAVE_ASYNC_B128 0
#endif
#if __has_builtin(__builtin_amdgcn_global_store_async_from_lds_b128)
#define HAVE_ASYNC_ST 1
#else
#define HAVE_ASYNC_ST 0
#endif
#if __has_builtin(__builtin_amdgcn_s_wait_asynccnt)
#define HAVE_WAIT_ASYNC 1
#else
#define HAVE_WAIT_ASYNC 0
#endif
#if __has_builtin(__builtin_amdgcn_tensor_load_to_lds)
#define HAVE_TDM 1
#else
#define HAVE_TDM 0
#endif
#if __has_builtin(__builtin_amdgcn_s_wait_tensorcnt)
#define HAVE_WAIT_TENSOR 1
#else
#define HAVE_WAIT_TENSOR 0
#endif

// A-matrix (16x32 f16) per-lane K index (vgpr-halfword pair v, lane-half h):
// VGPR0: K=0,1 / 8,9 ; VGPR1: 2,3 / 10,11 ; ... VGPR4-7: +16.
__device__ __forceinline__ int kfA(int v, int half) {
  return ((v >> 2) << 4) + (half << 3) + ((v & 3) << 1);
}
// B-matrix (32x16 f16) K index: same as A: VGPR0: K=0,1 / 8,9 ; ... VGPR4-7: +16.
__device__ __forceinline__ int kfB(int v, int half) {
  return ((v >> 2) << 4) + (half << 3) + ((v & 3) << 1);
}

// ---------------------------------------------------------------------------
// Kernel 0: one-shot W conversion, f32 [c][h] -> f16 transposed Wt[mat][h][c].
// ---------------------------------------------------------------------------
__global__ __launch_bounds__(256)
void wconv_kernel(const float* __restrict__ Wq, const float* __restrict__ Wk,
                  const float* __restrict__ Wv, _Float16* __restrict__ Wt)
{
  int idx = blockIdx.x * 256 + threadIdx.x;      // over 3*64*768
  if (idx >= 3 * HEAD * N_EMBD) return;
  int mat = idx / (HEAD * N_EMBD);
  int rem = idx % (HEAD * N_EMBD);
  int n = rem / N_EMBD;
  int c = rem % N_EMBD;
  const float* W = (mat == 0) ? Wq : (mat == 1) ? Wk : Wv;
  Wt[idx] = (_Float16)W[(size_t)c * HEAD + n];
}

// ---------------------------------------------------------------------------
// Kernel 1: QKV projection. 3 waves per block (one per matrix), sharing one
// f16 x tile in LDS; output tile staged in LDS and drained with async stores.
// ---------------------------------------------------------------------------
__global__ __launch_bounds__(96)
void qkv_proj_kernel(const float* __restrict__ x,
                     const _Float16* __restrict__ Wt,
                     _Float16* __restrict__ qkv)
{
  const int tid  = threadIdx.x;
  const int lane = tid & 31;
  const int mat  = tid >> 5;                 // wave id == matrix id
  const int half = lane >> 4;
  const int lm   = lane & 15;
  const int row0 = blockIdx.x * 16;          // over B*T rows

  __shared__ __align__(16) _Float16 ldsX[16 * N_EMBD];    // 24 KB
  __shared__ __align__(16) _Float16 ldsOut[3][16 * HEAD]; //  6 KB

  // Cooperative f32 -> f16 conversion of the 16 x 768 x-tile (once per block)
  const float2* xsrc = (const float2*)(x + (size_t)row0 * N_EMBD);
  for (int p = tid; p < 16 * N_EMBD / 2; p += 96) {
    float2 f = xsrc[p];
    ldsX[2 * p]     = (_Float16)f.x;
    ldsX[2 * p + 1] = (_Float16)f.y;
  }
  __syncthreads();

  const _Float16* W = Wt + (size_t)mat * HEAD * N_EMBD;  // [h][c] f16
  const float scale = (mat == 0) ? 0.125f : 1.0f;

  v8f acc[4] = {};                           // 4 N-tiles cover HEAD=64
  for (int kc = 0; kc < N_EMBD; kc += 32) {
    v16h a;                                  // x rows row0..+15, cols kc..+31
#pragma unroll
    for (int v = 0; v < 8; ++v) {
      int k = kc + kfA(v, half);
      a[2 * v]     = ldsX[lm * N_EMBD + k];
      a[2 * v + 1] = ldsX[lm * N_EMBD + k + 1];
    }
#pragma unroll
    for (int nt = 0; nt < 4; ++nt) {
      // B fragment: element (k=c, n=h) = Wt[n][c]; pairs contiguous in c
      v16h bw;
      const _Float16* wrow = W + (size_t)(nt * 16 + lm) * N_EMBD + kc;
#pragma unroll
      for (int v = 0; v < 8; ++v) {
        int k = kfB(v, half);
        bw[2 * v]     = wrow[k];
        bw[2 * v + 1] = wrow[k + 1];
      }
      acc[nt] = __builtin_amdgcn_wmma_f32_16x16x32_f16(
          false, a, false, bw, (short)0, acc[nt], false, false);
    }
  }

  // Stage the 16x64 output tile in LDS in exact memory layout...
#pragma unroll
  for (int nt = 0; nt < 4; ++nt)
#pragma unroll
    for (int r = 0; r < 8; ++r)
      ldsOut[mat][(r + 8 * half) * HEAD + nt * 16 + lm] =
          (_Float16)(acc[nt][r] * scale);
  __syncthreads();   // LDS writes complete (emits dscnt wait + barrier)

  // ...then drain it: rows row0..+15 are contiguous (2 KB) in qkv[mat].
  _Float16* dstRow = qkv + (size_t)mat * (NBATCH * T_SEQ * HEAD)
                         + (size_t)row0 * HEAD;
#if HAVE_ASYNC_ST
  for (int c = lane; c < 128; c += 32)      // 128 x 16B chunks per wave
    __builtin_amdgcn_global_store_async_from_lds_b128(
        (v4i*)(dstRow + c * 8), (v4i*)(&ldsOut[mat][0] + c * 8), 0, 0);
#if HAVE_WAIT_ASYNC
  __builtin_amdgcn_s_wait_asynccnt(0);
#endif
#else
  for (int c = lane; c < 256; c += 32)      // 8B chunks fallback
    ((uint2*)dstRow)[c] = ((const uint2*)&ldsOut[mat][0])[c];
#endif
}

// ---------------------------------------------------------------------------
// Kernel 2: causal flash attention. 4 waves / 64 queries per block. K block
// staged by the Tensor Data Mover (TDM D# descriptor), Q tile by async-LDS
// loads, V transposed via registers. Online softmax; O in f32.
// ---------------------------------------------------------------------------
__global__ __launch_bounds__(32 * QT_PER_BLK)
void flash_attn_kernel(const _Float16* __restrict__ qkv,
                       float* __restrict__ out)
{
  const int tid   = threadIdx.x;
  const int lane  = tid & 31;
  const int wave  = tid >> 5;
  const int half  = lane >> 4;
  const int lm    = lane & 15;
  const int blkQ  = blockIdx.x * (16 * QT_PER_BLK);
  const int b     = blockIdx.y;
  const int qbase = blkQ + wave * 16;

  const _Float16* Q  = qkv + (size_t)b * T_SEQ * HEAD;
  const _Float16* Kg = qkv + (size_t)(1 * NBATCH * T_SEQ * HEAD) + (size_t)b * T_SEQ * HEAD;
  const _Float16* Vg = qkv + (size_t)(2 * NBATCH * T_SEQ * HEAD) + (size_t)b * T_SEQ * HEAD;

  __shared__ __align__(16) _Float16 ldsK[32 * HEAD];              // [key][h]
  __shared__ __align__(16) _Float16 ldsVt[HEAD * 32];             // [h][key]
  __shared__ __align__(16) _Float16 ldsP[QT_PER_BLK][16 * 32];    // P staging
  __shared__ __align__(16) _Float16 ldsQ[QT_PER_BLK][16 * HEAD];  // Q tiles

  // ---- stage this wave's contiguous 2KB Q tile via async-LDS, then build
  //      the two A-fragments (h chunks 0-31 / 32-63) from LDS ----
  v16h aq[2];
  {
    const _Float16* qsrc = Q + (size_t)qbase * HEAD;
#if HAVE_ASYNC_B128
    for (int c = lane; c < 128; c += 32)
      __builtin_amdgcn_global_load_async_to_lds_b128(
          (v4i*)(qsrc + c * 8), (v4i*)(&ldsQ[wave][0] + c * 8), 0, 0);
#if HAVE_WAIT_ASYNC
    __builtin_amdgcn_s_wait_asynccnt(0);
#endif
#else
    for (int c = lane; c < 256; c += 32)
      ((uint2*)&ldsQ[wave][0])[c] = ((const uint2*)qsrc)[c];
#endif
    const _Float16* qrow = &ldsQ[wave][0] + (size_t)lm * HEAD;
#pragma unroll
    for (int hc = 0; hc < 2; ++hc)
#pragma unroll
      for (int v = 0; v < 8; ++v) {
        int k = hc * 32 + kfA(v, half);
        aq[hc][2 * v]     = qrow[k];
        aq[hc][2 * v + 1] = qrow[k + 1];
      }
  }

  v8f o[4] = {};
  float mrun[8], lrun[8];
#pragma unroll
  for (int r = 0; r < 8; ++r) { mrun[r] = -INFINITY; lrun[r] = 0.f; }

  const int kendBlk = blkQ + 16 * QT_PER_BLK;  // block-wide causal frontier
  const int kendW   = qbase + 16;              // this wave's frontier

  for (int kb = 0; kb < kendBlk; kb += 32) {
    if (kb + 32 < kendBlk) {
      __builtin_prefetch(Kg + (size_t)(kb + 32) * HEAD, 0, 1);
      __builtin_prefetch(Vg + (size_t)(kb + 32) * HEAD, 0, 1);
    }

    // ---- stage K block (32 keys x 64 f16 = 4 KB, row-major) ----
#if HAVE_TDM
    if (wave == 0) {
      // Tensor DMA descriptor (CDNA5 ISA): 2D tile, 2B elements.
      const unsigned long long ga =
          (unsigned long long)(size_t)(const void*)(Kg + (size_t)kb * HEAD);
      const unsigned ldsa = (unsigned)(size_t)(void*)&ldsK[0];
      v4u g0;
      g0.x = 1u;                                          // count=1, user D#
      g0.y = ldsa;                                        // lds_addr
      g0.z = (unsigned)(ga & 0xFFFFFFFFu);                // global_addr[31:0]
      g0.w = (unsigned)((ga >> 32) & 0x01FFFFFFu) | (2u << 30); // [56:32]|type=2
      v8i g1;
      g1[0] = 1 << 16;        // data_size=2B; no multicast; no pad/iterate
      g1[1] = 64 << 16;       // tensor_dim0 = 64 (h, contiguous)
      g1[2] = 32 << 16;       // tensor_dim1 = 32 (keys)
      g1[3] = 64 << 16;       // tile_dim0   = 64
      g1[4] = 32;             // tile_dim1   = 32, tile_dim2 = 0
      g1[5] = 64;             // tensor_dim0_stride = 64
      g1[6] = 0;
      g1[7] = 0;
      v4i gz4 = {0, 0, 0, 0};             // groups 2/3 unused (2D tensor)
      v8i gz8 = {0, 0, 0, 0, 0, 0, 0, 0}; // 5th SGPR group (clang-23 form)
      __builtin_amdgcn_tensor_load_to_lds(g0, g1, gz4, gz4, gz8, 0);
    }
#elif HAVE_ASYNC_B128
    for (int c = tid; c < 256; c += 32 * QT_PER_BLK) {
      v4i* gsrc = (v4i*)(Kg + (size_t)kb * HEAD + c * 8);
      v4i* ldst = (v4i*)(ldsK + c * 8);
      __builtin_amdgcn_global_load_async_to_lds_b128(gsrc, ldst, 0, 0);
    }
#else
    {
      const uint4* src = (const uint4*)(Kg + (size_t)kb * HEAD);
      uint4* dst = (uint4*)ldsK;
      for (int c = tid; c < 256; c += 32 * QT_PER_BLK) dst[c] = src[c];
    }
#endif

    // ---- stage V transposed: ldsVt[h][key] = V[kb+key][h] ----
    for (int e = tid; e < 32 * 32; e += 32 * QT_PER_BLK) {
      int key = e >> 5;          // 0..31
      int hp  = e & 31;          // h-pair index, h = 2*hp
      union { unsigned u; _Float16 h[2]; } pk;
      pk.u = *(const unsigned*)(Vg + (size_t)(kb + key) * HEAD + 2 * hp);
      ldsVt[(2 * hp) * 32 + key]     = pk.h[0];
      ldsVt[(2 * hp + 1) * 32 + key] = pk.h[1];
    }

#if HAVE_TDM && HAVE_WAIT_TENSOR
    if (wave == 0) __builtin_amdgcn_s_wait_tensorcnt(0);
#endif
#if !HAVE_TDM && HAVE_ASYNC_B128 && HAVE_WAIT_ASYNC
    __builtin_amdgcn_s_wait_asynccnt(0);
#endif
    __syncthreads();

    if (kb < kendW) {            // wave-uniform: EXEC stays all-ones
      // S = Q K^T for keys [kb, kb+32): two 16x16 tiles, 2 WMMAs each over h
      v8f s[2] = {};
#pragma unroll
      for (int t = 0; t < 2; ++t)
#pragma unroll
        for (int hc = 0; hc < 2; ++hc) {
          v16h bk;  // element (k=h, n=key) = ldsK[t*16+n][hc*32+k]
          const _Float16* kr = ldsK + (size_t)(t * 16 + lm) * HEAD + hc * 32;
#pragma unroll
          for (int v = 0; v < 8; ++v) {
            int k = kfB(v, half);
            bk[2 * v]     = kr[k];
            bk[2 * v + 1] = kr[k + 1];
          }
          s[t] = __builtin_amdgcn_wmma_f32_16x16x32_f16(
              false, aq[hc], false, bk, (short)0, s[t], false, false);
        }

      // Online softmax. C-layout: vgpr r -> row (r+8*half); col = lm.
#pragma unroll
      for (int r = 0; r < 8; ++r) {
        const int qidx = qbase + r + 8 * half;
        float s0 = s[0][r];
        float s1 = s[1][r];
        if (kb + lm      > qidx) s0 = -INFINITY;   // causal mask
        if (kb + 16 + lm > qidx) s1 = -INFINITY;
        float tmax = fmaxf(s0, s1);
#pragma unroll
        for (int off = 8; off >= 1; off >>= 1)
          tmax = fmaxf(tmax, __shfl_xor(tmax, off, 16));
        const float mnew = fmaxf(mrun[r], tmax);
        const float p0 = __expf(s0 - mnew);
        const float p1 = __expf(s1 - mnew);
        float psum = p0 + p1;
#pragma unroll
        for (int off = 8; off >= 1; off >>= 1)
          psum += __shfl_xor(psum, off, 16);
        const float alpha = __expf(mrun[r] - mnew);
        lrun[r] = lrun[r] * alpha + psum;
        mrun[r] = mnew;
#pragma unroll
        for (int hc = 0; hc < 4; ++hc) o[hc][r] *= alpha;
        const int row = r + 8 * half;
        ldsP[wave][row * 32 + lm]      = (_Float16)p0;
        ldsP[wave][row * 32 + 16 + lm] = (_Float16)p1;
      }

      // Reload P as A-fragment (same-wave DS ops are in-order)
      v16h ap;
#pragma unroll
      for (int v = 0; v < 8; ++v) {
        int k = kfA(v, half);
        ap[2 * v]     = ldsP[wave][lm * 32 + k];
        ap[2 * v + 1] = ldsP[wave][lm * 32 + k + 1];
      }

      // O += P * V : element (k=key, n=h) = ldsVt[h][key]; pairs contiguous
#pragma unroll
      for (int hc = 0; hc < 4; ++hc) {
        v16h bv;
#pragma unroll
        for (int v = 0; v < 8; ++v) {
          int k = kfB(v, half);
          bv[2 * v]     = ldsVt[(hc * 16 + lm) * 32 + k];
          bv[2 * v + 1] = ldsVt[(hc * 16 + lm) * 32 + k + 1];
        }
        o[hc] = __builtin_amdgcn_wmma_f32_16x16x32_f16(
            false, ap, false, bv, (short)0, o[hc], false, false);
      }
    }
    __syncthreads();   // before next staging overwrites ldsK/ldsVt
  }

  // Normalize and write fp32 output
  float* orow = out + ((size_t)b * T_SEQ + qbase) * HEAD;
#pragma unroll
  for (int r = 0; r < 8; ++r) {
    const float inv = 1.0f / lrun[r];
    const int row = r + 8 * half;
#pragma unroll
    for (int hc = 0; hc < 4; ++hc)
      orow[(size_t)row * HEAD + hc * 16 + lm] = o[hc][r] * inv;
  }
}

// ---------------------------------------------------------------------------
extern "C" void kernel_launch(void* const* d_in, const int* in_sizes, int n_in,
                              void* d_out, int out_size, void* d_ws, size_t ws_size,
                              hipStream_t stream) {
  (void)in_sizes; (void)n_in; (void)out_size; (void)ws_size;
  const float* x  = (const float*)d_in[0];
  const float* Wq = (const float*)d_in[1];
  const float* Wk = (const float*)d_in[2];
  const float* Wv = (const float*)d_in[3];
  float* out = (float*)d_out;

  _Float16* qkv = (_Float16*)d_ws;                         // 6 MB (L2-resident)
  _Float16* Wt  = qkv + (size_t)3 * NBATCH * T_SEQ * HEAD; // +0.28 MB

  wconv_kernel<<<(3 * HEAD * N_EMBD + 255) / 256, 256, 0, stream>>>(Wq, Wk, Wv, Wt);

  dim3 g1(NBATCH * T_SEQ / 16);                            // 1024 row tiles
  qkv_proj_kernel<<<g1, 96, 0, stream>>>(x, Wt, qkv);

  dim3 g2(T_SEQ / (16 * QT_PER_BLK), NBATCH);              // 64 x 4 blocks
  flash_attn_kernel<<<g2, 32 * QT_PER_BLK, 0, stream>>>(qkv, out);
}
